// SSGr1_23983097381055
// MI455X (gfx1250) — matrix operand order repaired and checked by reading, not compiled, not run on hardware
//
#include <hip/hip_runtime.h>
#include <math.h>

typedef _Float16 h16;
typedef __attribute__((ext_vector_type(16))) _Float16 v16h;
typedef __attribute__((ext_vector_type(8)))  _Float16 v8h;
typedef __attribute__((ext_vector_type(8)))  float    v8f;
typedef unsigned int u32x4 __attribute__((ext_vector_type(4)));
typedef int          i32x4 __attribute__((ext_vector_type(4)));
typedef int          i32x8 __attribute__((ext_vector_type(8)));

#ifndef __has_builtin
#define __has_builtin(x) 0
#endif
#if __has_builtin(__builtin_amdgcn_tensor_load_to_lds) && \
    __has_builtin(__builtin_amdgcn_s_wait_tensorcnt)
#define USE_TDM 1
#else
#define USE_TDM 0
#endif

#define KT    64    // K chunk: 2 x wmma_16x16x32 per fragment set
#define MTILE 64    // Cout macro-tile
#define NTILE 128   // pixel macro-tile
#define LDK   72    // LDS row stride in halfs: 144 B (64 data + 4-DWORD TDM pad)

// ---------------------------------------------------------------------------
// Implicit-GEMM conv, f16 in / f16 or f32 out, f32 accumulate.
// Double-buffered LDS pipeline: stage(next) issued before compute(cur);
// A tile staged by TDM (tensor_load_to_lds + s_wait_tensorcnt), B tile by
// cooperative im2col gather with hoisted K-decomposition.
// 8 waves: wave w -> M tile (w&3), N tiles (w>>2)*64 + {0,16,32,48}.
// 8 v_wmma_f32_16x16x32_f16 per wave per K-chunk.  act: 0=none 1=silu 2=lrelu
// ---------------------------------------------------------------------------
__global__ __launch_bounds__(256)
void k_conv_wmma(const h16* __restrict__ X, const h16* __restrict__ Wt,
                 const float* __restrict__ Bi, h16* __restrict__ Yh,
                 float* __restrict__ Yf,
                 int Cin, int H, int Win, int Cout, int Ks, int stride, int pad,
                 int OH, int OW, int act)
{
    __shared__ __align__(16) h16 Al[2][MTILE][LDK];
    __shared__ __align__(16) h16 Bl[2][NTILE][LDK];

    const int tid  = threadIdx.x;
    const int wave = tid >> 5;
    const int lane = tid & 31;
    const int lr   = lane & 15;
    const int lh   = lane >> 4;
    const int b    = blockIdx.z;
    const int m0g  = blockIdx.y * MTILE;
    const int n0g  = blockIdx.x * NTILE;
    const int OHW  = OH * OW;
    const int Ktot = Cin * Ks * Ks;
    const int ks2  = Ks * Ks;
    const int m0   = (wave & 3) * 16;
    const int n0   = (wave >> 2) * 64;

    v8f acc[4] = {{}, {}, {}, {}};

    // ---- stage one K-chunk into LDS buffer `buf` (A via TDM, B via gather) --
    auto stage = [&](int buf, int k0) {
#if USE_TDM
        if (wave == 0) {
            unsigned lds = (unsigned)(uintptr_t)&Al[buf][0][0];
            unsigned long long ga =
                (unsigned long long)(uintptr_t)(Wt + (size_t)m0g * Ktot + k0);
            unsigned td0 = (unsigned)(Ktot - k0);   // remaining K  (OOB -> 0)
            unsigned td1 = (unsigned)(Cout - m0g);  // remaining rows
            u32x4 g0;
            g0[0] = 1u;                                   // count=1, no gather
            g0[1] = lds;                                  // lds_addr
            g0[2] = (unsigned)(ga & 0xFFFFFFFFu);         // global_addr[31:0]
            g0[3] = (unsigned)((ga >> 32) & 0x1FFFFFFu)   // global_addr[56:32]
                    | (2u << 30);                         // type=2 (image)
            i32x8 g1;
            // data_size=2B, pad_enable, pad_interval=32 DW, pad_amount=4 DW
            g1[0] = (int)((1u << 16) | (1u << 20) | (4u << 22) | (3u << 25));
            g1[1] = (int)((td0 & 0xFFFFu) << 16);                 // tensor_dim0 lo
            g1[2] = (int)(((td0 >> 16) & 0xFFFFu) | ((td1 & 0xFFFFu) << 16));
            g1[3] = (int)(((td1 >> 16) & 0xFFFFu) | (64u << 16)); // tile_dim0=64
            g1[4] = 64;                                           // tile_dim1=64
            g1[5] = Ktot;                                         // dim0_stride lo
            g1[6] = 0;
            g1[7] = 0;
            i32x4 z4 = {0, 0, 0, 0};
#if __clang_major__ >= 23
            i32x8 z8 = {0, 0, 0, 0, 0, 0, 0, 0};
            __builtin_amdgcn_tensor_load_to_lds(g0, g1, z4, z4, z8, 0);
#else
            __builtin_amdgcn_tensor_load_to_lds(g0, g1, z4, z4, 0);
#endif
        }
#else
        {
            int kkA = tid & 63;
            int kA  = k0 + kkA;
            bool kvA = (kA < Ktot);
            for (int m = tid >> 6; m < MTILE; m += 4) {
                h16 v = (h16)0.f;
                int mg = m0g + m;
                if (kvA && mg < Cout) v = Wt[(size_t)mg * Ktot + kA];
                Al[buf][m][kkA] = v;
            }
        }
#endif
        // B tile: im2col gather; (ci,r,c) hoisted (kk invariant per thread)
        int kkB = tid & 63;
        int k   = k0 + kkB;
        bool kv = (k < Ktot);
        int ci  = k / ks2;
        int rem = k - ci * ks2;
        int r   = rem / Ks, c = rem - r * Ks;
        const h16* Xb = X + ((size_t)b * Cin + ci) * H * Win;
        int n  = tid >> 6;                 // 0..3, step 4
        int p  = n0g + n;
        int oh = p / OW, ow = p - oh * OW;
        for (; n < NTILE; n += 4) {
            h16 v = (h16)0.f;
            if (kv && p < OHW) {
                int ih = oh * stride - pad + r;
                int iw = ow * stride - pad + c;
                if ((unsigned)ih < (unsigned)H && (unsigned)iw < (unsigned)Win)
                    v = Xb[(size_t)ih * Win + iw];
            }
            Bl[buf][n][kkB] = v;
            p += 4; ow += 4;
            if (ow >= OW) { ow -= OW; ++oh; }
        }
    };

    // ---- pipeline: prologue stage, then overlap stage(next) with compute ---
    int cur = 0;
    stage(0, 0);
#if USE_TDM
    if (wave == 0) __builtin_amdgcn_s_wait_tensorcnt((short)0);
#endif
    __syncthreads();

    for (int k0 = 0; k0 < Ktot; k0 += KT) {
        int nk0 = k0 + KT;
        if (nk0 < Ktot) stage(cur ^ 1, nk0);   // async: fills other buffer

        #pragma unroll
        for (int kk = 0; kk < 2; ++kk) {
            v8h alo = *(const v8h*)&Al[cur][m0 + lr][kk * 32 + lh * 8];
            v8h ahi = *(const v8h*)&Al[cur][m0 + lr][kk * 32 + lh * 8 + 16];
            v16h a  = __builtin_shufflevector(alo, ahi,
                          0,1,2,3,4,5,6,7,8,9,10,11,12,13,14,15);
            #pragma unroll
            for (int t = 0; t < 4; ++t) {
                int nt = n0 + t * 16 + lr;
                v8h bl = *(const v8h*)&Bl[cur][nt][kk * 32 + lh * 8];
                v8h bh = *(const v8h*)&Bl[cur][nt][kk * 32 + lh * 8 + 16];
                v16h bf = __builtin_shufflevector(bl, bh,
                              0,1,2,3,4,5,6,7,8,9,10,11,12,13,14,15);
                acc[t] = __builtin_amdgcn_wmma_f32_16x16x32_f16(
                             false, a, false, bf, (short)0, acc[t], false, false);
            }
        }
#if USE_TDM
        if (wave == 0) __builtin_amdgcn_s_wait_tensorcnt((short)0);
#endif
        __syncthreads();
        cur ^= 1;
    }

    // ---- epilogue: bias + activation, f16 or f32 store ----
    #pragma unroll
    for (int t = 0; t < 4; ++t) {
        int p = n0g + n0 + t * 16 + lr;
        if (p < OHW) {
            #pragma unroll
            for (int r = 0; r < 8; ++r) {
                int c = m0g + m0 + r + 8 * lh;
                if (c < Cout) {
                    float y = acc[t][r] + (Bi ? Bi[c] : 0.f);
                    if (act == 1)      y = y / (1.f + __expf(-y));
                    else if (act == 2) y = (y >= 0.f) ? y : 0.2f * y;
                    size_t o = ((size_t)b * Cout + c) * OHW + p;
                    if (Yf) Yf[o] = y;
                    else    Yh[o] = (h16)y;
                }
            }
        }
    }
}

// ---------------------------------------------------------------------------
// Pack / unpack
// ---------------------------------------------------------------------------
__global__ void k_pack(const float* __restrict__ X, h16* __restrict__ Y, int n)
{
    int i = blockIdx.x * 256 + threadIdx.x;
    if (i < n) Y[i] = (h16)X[i];
}
__global__ void k_unpack(const h16* __restrict__ X, float* __restrict__ Y, int n)
{
    int i = blockIdx.x * 256 + threadIdx.x;
    if (i < n) Y[i] = (float)X[i];
}

// ---------------------------------------------------------------------------
// GroupNorm (f16 in/out, f32 math), fused optional SiLU.
// ---------------------------------------------------------------------------
__global__ __launch_bounds__(256)
void k_groupnorm(const h16* __restrict__ X, const float* __restrict__ G,
                 const float* __restrict__ Bt, h16* __restrict__ Y,
                 int C, int HW, int groups, int act)
{
    int b   = blockIdx.x / groups;
    int grp = blockIdx.x % groups;
    int cpg = C / groups;
    int len = cpg * HW;
    const h16* xp = X + (size_t)(b * C + grp * cpg) * HW;
    h16*       yp = Y + (size_t)(b * C + grp * cpg) * HW;

    float s = 0.f, s2 = 0.f;
    for (int i = threadIdx.x; i < len; i += 256) {
        float v = (float)xp[i]; s += v; s2 += v * v;
    }
    __shared__ float sh1[256], sh2[256];
    sh1[threadIdx.x] = s; sh2[threadIdx.x] = s2;
    __syncthreads();
    for (int off = 128; off > 0; off >>= 1) {
        if (threadIdx.x < off) {
            sh1[threadIdx.x] += sh1[threadIdx.x + off];
            sh2[threadIdx.x] += sh2[threadIdx.x + off];
        }
        __syncthreads();
    }
    float mu  = sh1[0] / len;
    float var = sh2[0] / len - mu * mu;
    float inv = rsqrtf(var + 1e-5f);
    for (int i = threadIdx.x; i < len; i += 256) {
        int c = grp * cpg + i / HW;
        float v = ((float)xp[i] - mu) * inv * G[c] + Bt[c];
        if (act == 1) v = v / (1.f + __expf(-v));
        yp[i] = (h16)v;
    }
}

// ---------------------------------------------------------------------------
// Switch attention: out = identity + scale * einsum(xf, softmax_t(m/10))
// ---------------------------------------------------------------------------
__global__ void k_switch_blend(const h16* __restrict__ XF, const h16* __restrict__ L,
                               const h16* __restrict__ I, const float* __restrict__ Sc,
                               h16* __restrict__ O, int B, int T, int F, int HW)
{
    int i = blockIdx.x * blockDim.x + threadIdx.x;
    if (i >= B * HW) return;
    int b = i / HW, p = i - b * HW;
    float sc = Sc ? Sc[0] : 1.f;

    float e[8];
    float mx = -1e30f;
    for (int t = 0; t < T; ++t) {
        float l = (float)L[((size_t)b * T + t) * HW + p] * 0.1f;
        e[t] = l; mx = fmaxf(mx, l);
    }
    float sum = 0.f;
    for (int t = 0; t < T; ++t) { e[t] = __expf(e[t] - mx); sum += e[t]; }
    float rs = 1.f / sum;
    for (int f = 0; f < F; ++f) {
        float a = 0.f;
        for (int t = 0; t < T; ++t)
            a += (float)XF[(((size_t)b * T + t) * F + f) * HW + p] * e[t];
        size_t o = ((size_t)b * F + f) * HW + p;
        O[o] = (h16)((float)I[o] + a * rs * sc);
    }
}

// ---------------------------------------------------------------------------
// Elementwise helpers
// ---------------------------------------------------------------------------
__global__ void k_imggrad(const float* __restrict__ X, h16* __restrict__ Y,
                          int NC, int H, int W)
{
    int i = blockIdx.x * 256 + threadIdx.x;
    int tot = NC * H * W;
    if (i >= tot) return;
    int w = i % W; int t = i / W; int h = t % H; int nc = t / H;
    const float* x = X + (size_t)nc * H * W;
    float up = (h > 0)     ? x[(h - 1) * W + w] : 0.f;
    float dn = (h < H - 1) ? x[(h + 1) * W + w] : 0.f;
    float lf = (w > 0)     ? x[h * W + w - 1]   : 0.f;
    float rt = (w < W - 1) ? x[h * W + w + 1]   : 0.f;
    float gv = dn - up, gh = rt - lf;
    Y[i] = (h16)sqrtf(gv * gv + gh * gh + 1e-6f);
}

__global__ void k_up2(const h16* __restrict__ X, h16* __restrict__ Y,
                      int NC, int H, int W)
{
    int i = blockIdx.x * 256 + threadIdx.x;
    int OH = 2 * H, OW = 2 * W;
    int tot = NC * OH * OW;
    if (i >= tot) return;
    int ow = i % OW; int t = i / OW; int oh = t % OH; int nc = t / OH;
    Y[i] = X[((size_t)nc * H + (oh >> 1)) * W + (ow >> 1)];
}

__global__ void k_add(const h16* __restrict__ A, const h16* __restrict__ B,
                      h16* __restrict__ O, int n)
{
    int i = blockIdx.x * 256 + threadIdx.x;
    if (i < n) O[i] = (h16)((float)A[i] + (float)B[i]);
}

__global__ void k_scale_bias(h16* __restrict__ X, const float* __restrict__ S,
                             const float* __restrict__ B, int n)
{
    int i = blockIdx.x * 256 + threadIdx.x;
    if (i < n) X[i] = (h16)((float)X[i] * S[0] + B[0]);
}

// ---------------------------------------------------------------------------
// Orchestration
// ---------------------------------------------------------------------------
extern "C" void kernel_launch(void* const* d_in, const int* in_sizes, int n_in,
                              void* d_out, int out_size, void* d_ws, size_t ws_size,
                              hipStream_t stream)
{
    (void)in_sizes; (void)out_size; (void)ws_size;
    const float* x = (const float*)d_in[0];
    (void)d_in[1]; (void)d_in[2];

    int pi = 3;  // sequential cursor over flattened param leaves
    auto nextp = [&]() -> const float* {
        int idx = (pi < n_in) ? pi : (n_in - 1);
        ++pi;
        return (const float*)d_in[idx];
    };

    h16* ws = (h16*)d_ws;
    size_t cur = 0;
    auto alloch = [&](size_t n) -> h16* {
        h16* p = ws + cur;
        cur += (n + 63) & ~(size_t)63;
        return p;
    };

    const int N = 4, NF = 64, H = 96, Wd = 96;
    const int HW = H * Wd;
    const size_t NB = (size_t)N * NF * HW;

    h16* wbuf = alloch(80 * 80 * 9 + 64);  // largest layer: 80 x (80*9)

    auto conv = [&](const h16* in, h16* outh, float* outf, int Cin, int Cout,
                    int Ks, int stride, bool bias, int act, int Hin, int Win) {
        const float* w  = nextp();
        const float* bi = bias ? nextp() : nullptr;
        int Kt = Cin * Ks * Ks;
        int nw = Cout * Kt;
        k_pack<<<(nw + 255) / 256, 256, 0, stream>>>(w, wbuf, nw);
        int OH = (Hin + stride - 1) / stride, OW = (Win + stride - 1) / stride;
        dim3 grid((OH * OW + NTILE - 1) / NTILE, (Cout + MTILE - 1) / MTILE, N);
        k_conv_wmma<<<grid, 256, 0, stream>>>(in, wbuf, bi, outh, outf,
                                              Cin, Hin, Win, Cout, Ks, stride,
                                              Ks / 2, OH, OW, act);
    };
    auto mcb_stack = [&](const h16* src, h16* dst, int T, h16* ta, h16* tb) {
        for (int t = 0; t < T; ++t) {
            conv(src, ta, nullptr, NF, 80, 3, 1, false, 2, H, Wd);
            conv(ta,  tb, nullptr, 80, 80, 3, 1, false, 2, H, Wd);
            conv(tb,  ta, nullptr, 80, 80, 3, 1, false, 2, H, Wd);
            conv(ta,  dst + (size_t)t * NB, nullptr, 80, NF, 3, 1, false, 0, H, Wd);
            const float* sc = nextp(); const float* bi = nextp();
            k_scale_bias<<<((int)NB + 255) / 256, 256, 0, stream>>>(
                dst + (size_t)t * NB, sc, bi, (int)NB);
        }
    };

    // workspace (f16 trunk)
    h16* xh    = alloch((size_t)N * 3 * HW);
    h16* xg    = alloch((size_t)N * 3 * HW);
    h16* feat  = alloch(NB);
    h16* xfs   = alloch(NB * 8);
    h16* t80a  = alloch((size_t)N * 80 * HW);
    h16* t80b  = alloch((size_t)N * 80 * HW);
    h16* lg    = alloch((size_t)N * 8 * HW);
    h16* x1    = alloch(NB);
    h16* gf    = alloch(NB);
    h16* bA    = alloch(NB);
    h16* bB    = alloch(NB);
    h16* g2    = alloch(NB);
    h16* up32a = alloch((size_t)N * 32 * HW * 4);
    h16* up32b = alloch((size_t)N * 32 * HW * 4);

    // 0) pack input, image gradient
    k_pack<<<(N * 3 * HW + 255) / 256, 256, 0, stream>>>(x, xh, N * 3 * HW);
    k_imggrad<<<(N * 3 * HW + 255) / 256, 256, 0, stream>>>(x, xg, N * 3, H, Wd);

    // 1) fea_conv (3->64, k3, bias)
    conv(xh, feat, nullptr, 3, NF, 3, 1, true, 0, H, Wd);

    // 2) switch 1
    mcb_stack(feat, xfs, 8, t80a, t80b);
    conv(feat, lg, nullptr, NF, 8, 1, 1, false, 0, H, Wd);
    {
        const float* sc = nextp();
        k_switch_blend<<<(N * HW + 255) / 256, 256, 0, stream>>>(
            xfs, lg, feat, sc, x1, N, 8, NF, HW);
    }

    // 3) gradient branch with reference join
    conv(xg, gf, nullptr, 3, NF, 3, 1, false, 0, H, Wd);
    k_add<<<((int)NB + 255) / 256, 256, 0, stream>>>(gf, x1, bA, (int)NB);
    conv(bA, bB, nullptr, NF, NF, 3, 1, false, 2, H, Wd);
    {
        const float* g = nextp(); const float* bt = nextp();
        k_groupnorm<<<N * 8, 256, 0, stream>>>(bB, g, bt, bA, NF, HW, 8, 1);
    }
    mcb_stack(bA, xfs, 4, t80a, t80b);
    conv(bA, lg, nullptr, NF, 4, 1, 1, false, 0, H, Wd);
    {
        const float* sc = nextp();
        k_switch_blend<<<(N * HW + 255) / 256, 256, 0, stream>>>(
            xfs, lg, gf, sc, bB, N, 4, NF, HW);
    }
    conv(bB, g2, nullptr, NF, NF, 3, 1, true, 2, H, Wd);   // grad_lr_conv

    // 4) grad upsample head -> go (fp32 out)
    float* out = (float*)d_out;
    const size_t GO = (size_t)N * 3 * 192 * 192;           // 442368
    conv(g2, up32a, nullptr, NF, 32, 1, 1, false, 0, H, Wd);
    k_up2<<<(N * 32 * HW * 4 + 255) / 256, 256, 0, stream>>>(up32a, up32b, N * 32, H, Wd);
    conv(up32b, up32a, nullptr, 32, 32, 3, 1, false, 2, 2 * H, 2 * Wd);
    conv(up32a, nullptr, out, 32, 3, 1, 1, true, 0, 2 * H, 2 * Wd);

    // 5) conjoin switch -> xo (fp32 out)
    k_add<<<((int)NB + 255) / 256, 256, 0, stream>>>(x1, g2, bA, (int)NB);
    conv(bA, bB, nullptr, NF, NF, 3, 1, false, 2, H, Wd);
    mcb_stack(bB, xfs, 8, t80a, t80b);
    conv(bB, lg, nullptr, NF, 8, 1, 1, false, 0, H, Wd);
    {
        const float* sc = nextp();
        k_switch_blend<<<(N * HW + 255) / 256, 256, 0, stream>>>(
            xfs, lg, x1, sc, bA, N, 8, NF, HW);
    }
    conv(bA, bB, nullptr, NF, NF, 3, 1, true, 2, H, Wd);   // final_lr_conv
    conv(bB, up32a, nullptr, NF, 32, 1, 1, false, 0, H, Wd);
    k_up2<<<(N * 32 * HW * 4 + 255) / 256, 256, 0, stream>>>(up32a, up32b, N * 32, H, Wd);
    conv(up32b, up32a, nullptr, 32, 32, 3, 1, true, 2, 2 * H, 2 * Wd);
    conv(up32a, nullptr, out + GO, 32, 3, 3, 1, false, 0, 2 * H, 2 * Wd);

    // 6) third output: grad feature g (f16 trunk -> fp32)
    k_unpack<<<((int)NB + 255) / 256, 256, 0, stream>>>(g2, out + 2 * GO, (int)NB);
}
